// gtaformerIteration_82652350644903
// MI455X (gfx1250) — matrix operand order repaired and checked by reading, not compiled, not run on hardware
//
#include <hip/hip_runtime.h>
#include <math.h>

#define LSEQ 192

typedef __attribute__((ext_vector_type(16))) _Float16 v16h;
typedef __attribute__((ext_vector_type(8)))  _Float16 v8h;
typedef __attribute__((ext_vector_type(8)))  float    v8f;

// concat two 8-float vectors (converted) into one 16-half fragment
__device__ __forceinline__ v16h f16cat(v8f lo, v8f hi) {
  v8h l = __builtin_convertvector(lo, v8h);
  v8h h = __builtin_convertvector(hi, v8h);
  return __builtin_shufflevector(l, h, 0,1,2,3,4,5,6,7,8,9,10,11,12,13,14,15);
}
// 16 contiguous halfs
__device__ __forceinline__ v16h ld16h(const _Float16* p) {
  v8h l = *(const v8h*)p;
  v8h h = *(const v8h*)(p + 8);
  return __builtin_shufflevector(l, h, 0,1,2,3,4,5,6,7,8,9,10,11,12,13,14,15);
}
#define WMMA(a, b, c) __builtin_amdgcn_wmma_f32_16x16x32_f16(false, (a), false, (b), (short)0, (c), false, false)

// Fragment geometry (wave32, 16x16x32 f16):
//  A: lane row = lane&15, K run  [k0+kh, +16), kh=(lane>>4)*16 (attn QK^T: A,B both [k0+h8,+8),[k0+16+h8,+8))
//  B: lane col = lane&15, K run  [k0+kh, +16), kh=(lane>>4)*16  (from f16 NxK)
//  C/D: col = lane&15, row r -> m = r + (lane>>4)*8

// ---------------- generic GEMM: C (=|+=) act(A*Bt^T + bias) ----------------
// Requires M%32==0, K%32==0, N%16==0 (weights zero-padded to N%16).
template<int ACT, bool ACCUM>
__global__ __launch_bounds__(256)
void gemm_wmma_kernel(const float* __restrict__ A, int lda,
                      const _Float16* __restrict__ Bt,
                      const float* __restrict__ bias,
                      float* __restrict__ C, int ldc,
                      int M, int N, int K)
{
  const int wave = threadIdx.x >> 5, lane = threadIdx.x & 31;
  const int m0 = blockIdx.x * 128 + (wave & 3) * 32;
  const int n0 = blockIdx.y * 64 + (wave >> 2) * 32;
  if (m0 >= M || n0 >= N) return;                   // wave-uniform
  const int l15 = lane & 15, h8 = (lane >> 4) * 8, kh = (lane >> 4) * 16;
  const bool n1 = (n0 + 16) < N;
  const float* a0p = A + (size_t)(m0 + l15) * lda;
  const float* a1p = A + (size_t)(m0 + 16 + l15) * lda;
  const _Float16* b0p = Bt + (size_t)(n0 + l15) * K + kh;
  const _Float16* b1p = Bt + (size_t)(n0 + 16 + l15) * K + kh;
  v8f acc00 = {}, acc01 = {}, acc10 = {}, acc11 = {};
  for (int k0 = 0; k0 < K; k0 += 32) {
    v16h a0 = f16cat(*(const v8f*)(a0p + k0 + kh), *(const v8f*)(a0p + k0 + kh + 8));
    v16h a1 = f16cat(*(const v8f*)(a1p + k0 + kh), *(const v8f*)(a1p + k0 + kh + 8));
    v16h b0 = ld16h(b0p + k0);
    acc00 = WMMA(a0, b0, acc00);
    acc10 = WMMA(a1, b0, acc10);
    if (n1) {
      v16h b1 = ld16h(b1p + k0);
      acc01 = WMMA(a0, b1, acc01);
      acc11 = WMMA(a1, b1, acc11);
    }
  }
  const int cn0 = n0 + l15, cn1 = n0 + 16 + l15;
  const float bv0 = bias ? bias[cn0] : 0.f;
  const float bv1 = (bias && n1) ? bias[cn1] : 0.f;
#pragma unroll
  for (int r = 0; r < 8; ++r) {
    const int mr0 = m0 + r + (lane >> 4) * 8, mr1 = mr0 + 16;
    float v00 = acc00[r] + bv0, v10 = acc10[r] + bv0;
    float v01 = acc01[r] + bv1, v11 = acc11[r] + bv1;
    if (ACT == 1) { v00 = fmaxf(v00, 0.f); v10 = fmaxf(v10, 0.f);
                    v01 = fmaxf(v01, 0.f); v11 = fmaxf(v11, 0.f); }
    if (ACT == 2) { v00 = 1.f/(1.f+__expf(-v00)); v10 = 1.f/(1.f+__expf(-v10));
                    v01 = 1.f/(1.f+__expf(-v01)); v11 = 1.f/(1.f+__expf(-v11)); }
    if (ACCUM) {
      C[(size_t)mr0 * ldc + cn0] += v00; C[(size_t)mr1 * ldc + cn0] += v10;
      if (n1) { C[(size_t)mr0 * ldc + cn1] += v01; C[(size_t)mr1 * ldc + cn1] += v11; }
    } else {
      C[(size_t)mr0 * ldc + cn0] = v00; C[(size_t)mr1 * ldc + cn0] = v10;
      if (n1) { C[(size_t)mr0 * ldc + cn1] = v01; C[(size_t)mr1 * ldc + cn1] = v11; }
    }
  }
}

// ---------------- fused attention: one wave per (s, h, q-tile) -------------
// Q,K: f32 (S*L, C) row-major (contraction dim contiguous). Vt: f16 (C, S*L).
// bias: f32 (L, L, 16) [stride 16, heads 0..H-1 valid].
__global__ __launch_bounds__(32)
void attn_fused_kernel(const float* __restrict__ Q, const float* __restrict__ Km,
                       const _Float16* __restrict__ Vt, const float* __restrict__ bias,
                       const float* __restrict__ mask, float* __restrict__ O,
                       int H, int dh, int C, int Rv, float scale)
{
  __shared__ float sm[16 * LSEQ];
  const int lane = threadIdx.x & 31;
  const int nqt = LSEQ / 16;
  const int qt = blockIdx.x % nqt;
  const int h  = (blockIdx.x / nqt) % H;
  const int s  = blockIdx.x / (nqt * H);
  const int q0 = qt * 16;
  const int l15 = lane & 15, h8 = (lane >> 4) * 8, kh = (lane >> 4) * 16;
  const float* Qb = Q + (size_t)s * LSEQ * C + h * dh;
  const float* Kb = Km + (size_t)s * LSEQ * C + h * dh;
  const _Float16* Vb = Vt + (size_t)h * dh * Rv + (size_t)s * LSEQ;
  const v8f zv = {};

  // pass 1: logits = scale*Q K^T + bias, masked
  const float* qrow = Qb + (size_t)(q0 + l15) * C;
  for (int kt = 0; kt < LSEQ; kt += 16) {
    const float* krow = Kb + (size_t)(kt + l15) * C;
    v8f acc = {};
    for (int d0 = 0; d0 < dh; d0 += 32) {
      v8f q0v = *(const v8f*)(qrow + d0 + h8) * scale;
      v8f q1v = (d0 + 16 < dh) ? *(const v8f*)(qrow + d0 + h8 + 16) * scale : zv;
      v16h a = f16cat(q0v, q1v);
      v8f k0v = zv, k1v = zv;
      k0v = *(const v8f*)(krow + d0 + h8);
      if (d0 + 16 < dh) {
        k1v = *(const v8f*)(krow + d0 + h8 + 16);
      }
      v16h b = f16cat(k0v, k1v);
      acc = WMMA(a, b, acc);
    }
#pragma unroll
    for (int r = 0; r < 8; ++r) {
      int m = r + (lane >> 4) * 8;
      int kk = kt + l15;
      float v = acc[r] + bias[((size_t)(q0 + m) * LSEQ + kk) * 16 + h];
      if (mask[kk] <= 0.f) v = -1e30f;
      sm[m * LSEQ + kk] = v;
    }
  }
  __syncthreads();
  // softmax: one lane per q-row
  if (lane < 16) {
    float mx = -1e30f;
    for (int j = 0; j < LSEQ; ++j) mx = fmaxf(mx, sm[lane * LSEQ + j]);
    float sum = 0.f;
    for (int j = 0; j < LSEQ; ++j) {
      float e = __expf(sm[lane * LSEQ + j] - mx);
      sm[lane * LSEQ + j] = e; sum += e;
    }
    float inv = 1.f / sum;
    for (int j = 0; j < LSEQ; ++j) sm[lane * LSEQ + j] *= inv;
  }
  __syncthreads();
  // pass 2: O = P V
  const float* smrow = sm + l15 * LSEQ;
  for (int d0 = 0; d0 < dh; d0 += 16) {
    const _Float16* vrow = Vb + (size_t)(d0 + l15) * Rv;
    v8f acc = {};
    for (int k0 = 0; k0 < LSEQ; k0 += 32) {
      v16h a = f16cat(*(const v8f*)(smrow + k0 + kh), *(const v8f*)(smrow + k0 + kh + 8));
      v16h b = ld16h(vrow + k0 + kh);
      acc = WMMA(a, b, acc);
    }
#pragma unroll
    for (int r = 0; r < 8; ++r) {
      int m = q0 + r + (lane >> 4) * 8;
      int n = d0 + l15;
      O[(size_t)(s * LSEQ + m) * C + h * dh + n] = acc[r];
    }
  }
}

// ------------- triangle einsum: Out[i,j,c] = sum_k Lt[c,i,k]*Rt[c,j,k] -----
__global__ __launch_bounds__(256)
void tri_einsum_kernel(const float* __restrict__ Lt, const float* __restrict__ Rt,
                       float* __restrict__ O, int C)
{
  const int wave = threadIdx.x >> 5, lane = threadIdx.x & 31;
  const int t32 = LSEQ / 32, per_c = t32 * t32;        // 6, 36
  int wid = blockIdx.x * 8 + wave;
  if (wid >= C * per_c) return;
  int c = wid / per_c, rem = wid % per_c;
  int i0 = (rem / t32) * 32, j0 = (rem % t32) * 32;
  const int l15 = lane & 15, h8 = (lane >> 4) * 8, kh = (lane >> 4) * 16;
  const float* Lc = Lt + (size_t)c * LSEQ * LSEQ;
  const float* Rc = Rt + (size_t)c * LSEQ * LSEQ;
  const float* a0p = Lc + (size_t)(i0 + l15) * LSEQ;
  const float* a1p = Lc + (size_t)(i0 + 16 + l15) * LSEQ;
  const float* b0p = Rc + (size_t)(j0 + l15) * LSEQ + kh;
  const float* b1p = Rc + (size_t)(j0 + 16 + l15) * LSEQ + kh;
  v8f acc00 = {}, acc01 = {}, acc10 = {}, acc11 = {};
  for (int k0 = 0; k0 < LSEQ; k0 += 32) {
    v16h a0 = f16cat(*(const v8f*)(a0p + k0 + kh), *(const v8f*)(a0p + k0 + kh + 8));
    v16h a1 = f16cat(*(const v8f*)(a1p + k0 + kh), *(const v8f*)(a1p + k0 + kh + 8));
    v16h b0 = f16cat(*(const v8f*)(b0p + k0), *(const v8f*)(b0p + k0 + 8));
    v16h b1 = f16cat(*(const v8f*)(b1p + k0), *(const v8f*)(b1p + k0 + 8));
    acc00 = WMMA(a0, b0, acc00);
    acc01 = WMMA(a0, b1, acc01);
    acc10 = WMMA(a1, b0, acc10);
    acc11 = WMMA(a1, b1, acc11);
  }
#pragma unroll
  for (int r = 0; r < 8; ++r) {
    int mr0 = i0 + r + (lane >> 4) * 8, mr1 = mr0 + 16;
    int nc0 = j0 + l15, nc1 = nc0 + 16;
    O[((size_t)mr0 * LSEQ + nc0) * C + c] = acc00[r];
    O[((size_t)mr0 * LSEQ + nc1) * C + c] = acc01[r];
    O[((size_t)mr1 * LSEQ + nc0) * C + c] = acc10[r];
    O[((size_t)mr1 * LSEQ + nc1) * C + c] = acc11[r];
  }
}

// ---------------- weight / activation conversion ---------------------------
// f32 (K,N) row-major -> f16 (Npad,K) transposed, zero-padded
__global__ void wcvt_kernel(const float* __restrict__ W, _Float16* __restrict__ Wt,
                            int K, int N, int Npad) {
  size_t tot = (size_t)Npad * K;
  for (size_t i = (size_t)blockIdx.x * blockDim.x + threadIdx.x; i < tot;
       i += (size_t)gridDim.x * blockDim.x) {
    int n = (int)(i / K), k = (int)(i % K);
    Wt[i] = (_Float16)((n < N) ? W[(size_t)k * N + n] : 0.f);
  }
}
// f32 (R,C) -> f16 (C,R)
__global__ void tcvt_kernel(const float* __restrict__ X, _Float16* __restrict__ Xt,
                            int R, int C) {
  size_t tot = (size_t)R * C;
  for (size_t i = (size_t)blockIdx.x * blockDim.x + threadIdx.x; i < tot;
       i += (size_t)gridDim.x * blockDim.x) {
    int r = (int)(i / C), c = (int)(i % C);
    Xt[(size_t)c * R + r] = (_Float16)X[i];
  }
}

// ---------------- LayerNorm over last dim (one wave per row) ---------------
__global__ __launch_bounds__(256)
void ln_kernel(const float* __restrict__ x, float* __restrict__ y,
               const float* __restrict__ g, const float* __restrict__ b,
               int R, int C)
{
  int row = blockIdx.x * (blockDim.x >> 5) + (threadIdx.x >> 5);
  int lane = threadIdx.x & 31;
  if (row >= R) return;
  const float* xr = x + (size_t)row * C;
  float s = 0.f;
  for (int c = lane; c < C; c += 32) s += xr[c];
  for (int off = 16; off; off >>= 1) s += __shfl_xor(s, off, 32);
  float mean = s / C;
  float v = 0.f;
  for (int c = lane; c < C; c += 32) { float d = xr[c] - mean; v += d * d; }
  for (int off = 16; off; off >>= 1) v += __shfl_xor(v, off, 32);
  float rstd = rsqrtf(v / C + 1e-5f);
  float* yr = y + (size_t)row * C;
  for (int c = lane; c < C; c += 32) yr[c] = (xr[c] - mean) * rstd * g[c] + b[c];
}

// ---------------- elementwise / reshape helpers ----------------------------
__global__ void ew_mul_kernel(float* d, const float* a, size_t n) {
  for (size_t i = (size_t)blockIdx.x * blockDim.x + threadIdx.x; i < n;
       i += (size_t)gridDim.x * blockDim.x) d[i] *= a[i];
}
__global__ void gate_accum_kernel(float* d, const float* u, const float* g, size_t n) {
  for (size_t i = (size_t)blockIdx.x * blockDim.x + threadIdx.x; i < n;
       i += (size_t)gridDim.x * blockDim.x) d[i] += u[i] * g[i];
}
__global__ void rowscale_kernel(float* d, const float* m, int R, int C) {
  size_t n = (size_t)R * C;
  for (size_t i = (size_t)blockIdx.x * blockDim.x + threadIdx.x; i < n;
       i += (size_t)gridDim.x * blockDim.x) d[i] *= m[i / C];
}
__global__ void maskpair_kernel(float* dst, const float* P, const float* G,
                                const float* mask, int L, int C) {
  size_t n = (size_t)L * L * C;
  for (size_t i = (size_t)blockIdx.x * blockDim.x + threadIdx.x; i < n;
       i += (size_t)gridDim.x * blockDim.x) {
    size_t rc = i / C;
    int y = (int)(rc % L), x = (int)(rc / L);
    dst[i] = mask[x] * mask[y] * P[i] * G[i];
  }
}
// (a,b,c) -> channel-major (c,a,b); swap: (c,b,a)
__global__ void ctrans_kernel(const float* __restrict__ in, float* __restrict__ out,
                              int L, int C, int swap) {
  size_t n = (size_t)L * L * C;
  for (size_t i = (size_t)blockIdx.x * blockDim.x + threadIdx.x; i < n;
       i += (size_t)gridDim.x * blockDim.x) {
    int c = (int)(i % C); size_t rc = i / C;
    int b = (int)(rc % L), a = (int)(rc / L);
    size_t dst = (size_t)c * L * L + (swap ? (size_t)b * L + a : (size_t)a * L + b);
    out[dst] = in[i];
  }
}
__global__ void ptrans_kernel(const float* __restrict__ in, float* __restrict__ out,
                              int L, int C) {
  size_t n = (size_t)L * L * C;
  for (size_t i = (size_t)blockIdx.x * blockDim.x + threadIdx.x; i < n;
       i += (size_t)gridDim.x * blockDim.x) {
    int c = (int)(i % C); size_t rc = i / C;
    int jj = (int)(rc % L), ii = (int)(rc / L);
    out[((size_t)jj * L + ii) * C + c] = in[i];
  }
}
__global__ void ptrans_accum_kernel(float* __restrict__ dst, const float* __restrict__ src,
                                    int L, int C) {
  size_t n = (size_t)L * L * C;
  for (size_t i = (size_t)blockIdx.x * blockDim.x + threadIdx.x; i < n;
       i += (size_t)gridDim.x * blockDim.x) {
    int c = (int)(i % C); size_t rc = i / C;
    int jj = (int)(rc % L), ii = (int)(rc / L);
    dst[i] += src[((size_t)jj * L + ii) * C + c];
  }
}
__global__ void opm_build_kernel(const float* __restrict__ lv, const float* __restrict__ rv,
                                 float* __restrict__ T, int L, int NOc) {
  size_t n = (size_t)L * L * 2 * NOc;
  for (size_t i = (size_t)blockIdx.x * blockDim.x + threadIdx.x; i < n;
       i += (size_t)gridDim.x * blockDim.x) {
    int t = (int)(i % (2 * NOc)); size_t pr = i / (2 * NOc);
    int j = (int)(pr % L), ii = (int)(pr / L);
    int tt = t % NOc;
    float a = lv[(size_t)j * NOc + tt], b = rv[(size_t)ii * NOc + tt];
    T[i] = (t < NOc) ? a * b : a - b;
  }
}
__global__ void mean_j_kernel(const float* __restrict__ H1, float* __restrict__ out,
                              int L, int F) {
  int i = blockIdx.x;
  for (int f = threadIdx.x; f < F; f += blockDim.x) {
    float s = 0.f;
    for (int j = 0; j < L; ++j) s += H1[((size_t)i * L + j) * F + f];
    out[(size_t)i * F + f] = s / L;
  }
}
__global__ void mean_i_kernel(const float* __restrict__ H1, float* __restrict__ out,
                              int L, int F) {
  int j = blockIdx.x;
  for (int f = threadIdx.x; f < F; f += blockDim.x) {
    float s = 0.f;
    for (int i = 0; i < L; ++i) s += H1[((size_t)i * L + j) * F + f];
    out[(size_t)j * F + f] = s / L;
  }
}

// ===========================================================================
extern "C" void kernel_launch(void* const* d_in, const int* in_sizes, int n_in,
                              void* d_out, int out_size, void* d_ws, size_t ws_size,
                              hipStream_t stream) {
  (void)in_sizes; (void)n_in; (void)out_size; (void)ws_size;
  const int L = 192, CS = 384, CP = 128, Hh = 8, NO = 32;
  const int L2 = L * L;
  auto f = [&](int i) { return (const float*)d_in[i]; };
  const float* seq_in  = f(0);
  const float* pair_in = f(1);
  const float* mask    = f(2);
  // params (jax tree-flatten order): 3..10 opm, 11..16 pair_trans,
  // 17..28 seq_attn, 29..34 seq_left, 35..40 seq_right, 41..46 seq_trans,
  // 47..56 ta_end, 57..66 ta_start, 67..82 tm_in, 83..98 tm_out

  float* seqO  = (float*)d_out;
  float* pairO = seqO + (size_t)L * CS;

  float* w = (float*)d_ws;
  const size_t PSZ = (size_t)L2 * CP;
  float* W0 = w;               float* W1 = W0 + PSZ; float* W2 = W1 + PSZ;
  float* W3 = W2 + PSZ;        float* W4 = W3 + PSZ; float* W5 = W4 + PSZ;
  float* W6 = W5 + PSZ;        float* W7 = W6 + PSZ;
  float* BIASB = W7 + PSZ;                    // L2*16  (head-padded bias)
  float* BF    = BIASB + (size_t)L2 * 16;     // L2*512
  float* s_    = BF + (size_t)L2 * 512;
  float* qS = s_ + (size_t)L * CS; float* kS = qS + (size_t)L * CS;
  float* vS = kS + (size_t)L * CS; float* gS = vS + (size_t)L * CS;
  float* oS = gS + (size_t)L * CS;
  float* t1 = oS + (size_t)L * CS;            // 192*1536
  float* lb = t1 + (size_t)L * 4 * CS;
  float* rb = lb + (size_t)L * NO;
  float* MB = rb + (size_t)L * NO;            // 192*512
  float* T64 = MB + (size_t)L * 512;          // L2*64
  _Float16* VT  = (_Float16*)(T64 + (size_t)L2 * 64);  // PSZ halfs
  _Float16* wh  = VT + PSZ;                   // f16 weight arena (bump alloc)

  dim3 ewg(2048), ewb(256);
  auto gemm = [&](const float* A, int lda, int widx, int K, int N,
                  const float* bias, float* C, int ldc, int M, int act, bool accum) {
    int Npad = (N + 15) & ~15;
    _Float16* Bt = wh; wh += (size_t)Npad * K;
    wcvt_kernel<<<dim3(512), ewb, 0, stream>>>(f(widx), Bt, K, N, Npad);
    dim3 g((M + 127) / 128, (Npad + 63) / 64), blk(256);
    if (!accum) {
      if (act == 0)      gemm_wmma_kernel<0, false><<<g, blk, 0, stream>>>(A, lda, Bt, bias, C, ldc, M, Npad, K);
      else if (act == 1) gemm_wmma_kernel<1, false><<<g, blk, 0, stream>>>(A, lda, Bt, bias, C, ldc, M, Npad, K);
      else               gemm_wmma_kernel<2, false><<<g, blk, 0, stream>>>(A, lda, Bt, bias, C, ldc, M, Npad, K);
    } else {
      if (act == 0)      gemm_wmma_kernel<0, true ><<<g, blk, 0, stream>>>(A, lda, Bt, bias, C, ldc, M, Npad, K);
      else if (act == 1) gemm_wmma_kernel<1, true ><<<g, blk, 0, stream>>>(A, lda, Bt, bias, C, ldc, M, Npad, K);
      else               gemm_wmma_kernel<2, true ><<<g, blk, 0, stream>>>(A, lda, Bt, bias, C, ldc, M, Npad, K);
    }
  };
  auto LN = [&](const float* x, float* y, const float* g, const float* b, int R, int C) {
    ln_kernel<<<dim3((R + 7) / 8), dim3(256), 0, stream>>>(x, y, g, b, R, C);
  };

  // residual bases
  hipMemcpyAsync(seqO, seq_in, sizeof(float) * (size_t)L * CS, hipMemcpyDeviceToDevice, stream);
  hipMemcpyAsync(pairO, pair_in, sizeof(float) * PSZ, hipMemcpyDeviceToDevice, stream);

  // ---- Stage 1: seq attention with pair bias ------------------------------
  LN(seqO, s_, f(27), f(26), L, CS);            // seq_ln
  LN(pairO, W0, f(25), f(24), L2, CP);          // pair_ln
  gemm(W0, CP, 28, CP, Hh, nullptr, BIASB, 16, L2, 0, false);   // bias (i,j,16)
  gemm(s_, CS, 22, CS, CS, nullptr, qS, CS, L, 0, false);       // wq
  gemm(s_, CS, 20, CS, CS, nullptr, kS, CS, L, 0, false);       // wk
  gemm(s_, CS, 23, CS, CS, nullptr, vS, CS, L, 0, false);       // wv
  gemm(s_, CS, 19, CS, CS, f(17), gS, CS, L, 2, false);         // sigmoid(wg+bg)
  tcvt_kernel<<<ewg, ewb, 0, stream>>>(vS, VT, L, CS);
  attn_fused_kernel<<<dim3(1 * Hh * (L / 16)), dim3(32), 0, stream>>>(
      qS, kS, VT, BIASB, mask, oS, Hh, 48, CS, L, 1.f / sqrtf(48.f));
  ew_mul_kernel<<<ewg, ewb, 0, stream>>>(oS, gS, (size_t)L * CS);
  gemm(oS, CS, 21, CS, CS, f(18), seqO, CS, L, 0, true);        // + wo,bo

  // ---- Stage 2: seq transition -------------------------------------------
  LN(seqO, s_, f(44), f(43), L, CS);
  gemm(s_, CS, 45, CS, 4 * CS, f(41), t1, 4 * CS, L, 1, false);
  gemm(t1, 4 * CS, 46, 4 * CS, CS, f(42), seqO, CS, L, 0, true);

  // ---- Stage 3: outer product (OPM) --------------------------------------
  LN(seqO, s_, f(7), f(6), L, CS);
  gemm(s_, CS, 8, CS, NO, f(3), lb, NO, L, 0, false);
  rowscale_kernel<<<ewg, ewb, 0, stream>>>(lb, mask, L, NO);
  gemm(s_, CS, 10, CS, NO, f(5), rb, NO, L, 0, false);
  rowscale_kernel<<<ewg, ewb, 0, stream>>>(rb, mask, L, NO);
  opm_build_kernel<<<ewg, ewb, 0, stream>>>(lb, rb, T64, L, NO);
  gemm(T64, 2 * NO, 9, 2 * NO, CP, f(4), pairO, CP, L2, 0, true);

  // ---- Stage 4: triangle multiplications ----------------------------------
  auto trimul = [&](int b0, int swap) {
    // b0: bfg,+1 bl,+2 blg,+3 bo,+4 br,+5 brg,+6 fln.b,+7 fln.g,+8 ln.b,+9 ln.g,
    //     +10 wfg,+11 wl,+12 wlg,+13 wo,+14 wr,+15 wrg
    LN(pairO, W0, f(b0 + 9), f(b0 + 8), L2, CP);
    gemm(W0, CP, b0 + 11, CP, CP, f(b0 + 1), W1, CP, L2, 0, false);
    gemm(W0, CP, b0 + 12, CP, CP, f(b0 + 2), W2, CP, L2, 2, false);
    maskpair_kernel<<<ewg, ewb, 0, stream>>>(W3, W1, W2, mask, L, CP);   // left
    gemm(W0, CP, b0 + 14, CP, CP, f(b0 + 4), W1, CP, L2, 0, false);
    gemm(W0, CP, b0 + 15, CP, CP, f(b0 + 5), W2, CP, L2, 2, false);
    maskpair_kernel<<<ewg, ewb, 0, stream>>>(W4, W1, W2, mask, L, CP);   // right
    ctrans_kernel<<<ewg, ewb, 0, stream>>>(W3, W5, L, CP, swap);
    ctrans_kernel<<<ewg, ewb, 0, stream>>>(W4, W6, L, CP, swap);
    tri_einsum_kernel<<<dim3((CP * 36 + 7) / 8), dim3(256), 0, stream>>>(W5, W6, W7, CP);
    LN(W7, W1, f(b0 + 7), f(b0 + 6), L2, CP);                            // fln
    gemm(W1, CP, b0 + 13, CP, CP, f(b0 + 3), W2, CP, L2, 0, false);      // U
    gemm(W0, CP, b0 + 10, CP, CP, f(b0 + 0), W5, CP, L2, 2, false);      // gate
    gate_accum_kernel<<<ewg, ewb, 0, stream>>>(pairO, W2, W5, PSZ);
  };
  trimul(83, 0);  // tm_out, per_row
  trimul(67, 1);  // tm_in, per_col

  // ---- Stage 5: triangle attentions ---------------------------------------
  auto triattn = [&](int a0, int lnb, int lng, int wp, bool percol) {
    // a0: bg,+1 bo,+2 wg,+3 wk,+4 wo,+5 wq,+6 wv
    const float* src = pairO;
    if (percol) { ptrans_kernel<<<ewg, ewb, 0, stream>>>(pairO, W6, L, CP); src = W6; }
    LN(src, W0, f(lng), f(lnb), L2, CP);
    gemm(W0, CP, wp, CP, Hh, nullptr, BIASB, 16, L2, 0, false);
    gemm(W0, CP, a0 + 5, CP, CP, nullptr, W1, CP, L2, 0, false);   // Q
    gemm(W0, CP, a0 + 3, CP, CP, nullptr, W2, CP, L2, 0, false);   // K
    gemm(W0, CP, a0 + 6, CP, CP, nullptr, W3, CP, L2, 0, false);   // V
    gemm(W0, CP, a0 + 2, CP, CP, f(a0 + 0), W4, CP, L2, 2, false); // gate
    tcvt_kernel<<<ewg, ewb, 0, stream>>>(W3, VT, L2, CP);
    attn_fused_kernel<<<dim3(L * Hh * (L / 16)), dim3(32), 0, stream>>>(
        W1, W2, VT, BIASB, mask, W5, Hh, 16, CP, L2, 0.25f);
    ew_mul_kernel<<<ewg, ewb, 0, stream>>>(W5, W4, PSZ);
    if (!percol) {
      gemm(W5, CP, a0 + 4, CP, CP, f(a0 + 1), pairO, CP, L2, 0, true);
    } else {
      gemm(W5, CP, a0 + 4, CP, CP, f(a0 + 1), W7, CP, L2, 0, false);
      ptrans_accum_kernel<<<ewg, ewb, 0, stream>>>(pairO, W7, L, CP);
    }
  };
  triattn(57, 64, 65, 66, false);  // ta_start
  triattn(47, 54, 55, 56, true);   // ta_end

  // ---- Stage 6: pair transition -------------------------------------------
  LN(pairO, W0, f(14), f(13), L2, CP);
  gemm(W0, CP, 15, CP, 512, f(11), BF, 512, L2, 1, false);
  gemm(BF, 512, 16, 512, CP, f(12), pairO, CP, L2, 0, true);

  // ---- Stage 7: seq updates (mean commutes with second linear) ------------
  LN(pairO, W0, f(32), f(31), L2, CP);                                  // seq_left
  gemm(W0, CP, 33, CP, 512, f(29), BF, 512, L2, 1, false);
  mean_j_kernel<<<dim3(L), dim3(256), 0, stream>>>(BF, MB, L, 512);
  gemm(MB, 512, 34, 512, CS, f(30), seqO, CS, L, 0, true);

  LN(pairO, W0, f(38), f(37), L2, CP);                                  // seq_right
  gemm(W0, CP, 39, CP, 512, f(35), BF, 512, L2, 1, false);
  mean_i_kernel<<<dim3(L), dim3(256), 0, stream>>>(BF, MB, L, 512);
  gemm(MB, 512, 40, 512, CS, f(36), seqO, CS, L, 0, true);
}
